// PSPDecoder_75196287419007
// MI455X (gfx1250) — matrix operand order repaired and checked by reading, not compiled, not run on hardware
//
#include <hip/hip_runtime.h>
#include <hip/hip_bf16.h>
#include <stdint.h>

// ---------------- types ----------------
typedef __attribute__((ext_vector_type(16))) __bf16 v16bf;
typedef __attribute__((ext_vector_type(8)))  float  v8f;
typedef unsigned int u32x4 __attribute__((ext_vector_type(4)));
typedef float        f32x4 __attribute__((ext_vector_type(4)));
typedef int          i32x4v __attribute__((vector_size(16)));   // matches async-LDS builtin param

union BF16x16 { v16bf v; unsigned short u[16]; u32x4 q[2]; };
union F32x8   { v8f  v; float f[8]; };

__device__ __forceinline__ unsigned short bf16_trunc(float f) {
    union { float f; unsigned u; } x; x.f = f;
    return (unsigned short)(x.u >> 16);
}
__device__ __forceinline__ float bf16_f(unsigned short h) {
    union { float f; unsigned u; } x; x.u = ((unsigned)h) << 16;
    return x.f;
}
__device__ __forceinline__ void split8(const float* v, u32x4& qh, u32x4& ql) {
    unsigned short h[8], l[8];
    #pragma unroll
    for (int d = 0; d < 8; ++d) {
        h[d] = bf16_trunc(v[d]);
        l[d] = bf16_trunc(v[d] - bf16_f(h[d]));
    }
    qh.x = (unsigned)h[0] | ((unsigned)h[1] << 16);
    qh.y = (unsigned)h[2] | ((unsigned)h[3] << 16);
    qh.z = (unsigned)h[4] | ((unsigned)h[5] << 16);
    qh.w = (unsigned)h[6] | ((unsigned)h[7] << 16);
    ql.x = (unsigned)l[0] | ((unsigned)l[1] << 16);
    ql.y = (unsigned)l[2] | ((unsigned)l[3] << 16);
    ql.z = (unsigned)l[4] | ((unsigned)l[5] << 16);
    ql.w = (unsigned)l[6] | ((unsigned)l[7] << 16);
}

// ---- gfx1250 async global->LDS copy (ASYNCcnt path), with safe fallback ----
#if defined(__HIP_DEVICE_COMPILE__) && __has_builtin(__builtin_amdgcn_global_load_async_to_lds_b128)
#define USE_ASYNC_LDS 1
#else
#define USE_ASYNC_LDS 0
#endif

#if USE_ASYNC_LDS
__device__ __forceinline__ void gl2lds16(const void* g, void* l) {
    __attribute__((address_space(1))) i32x4v* gp =
        (__attribute__((address_space(1))) i32x4v*)(unsigned long long)(uintptr_t)g;
    __attribute__((address_space(3))) i32x4v* lp =
        (__attribute__((address_space(3))) i32x4v*)(unsigned)(uintptr_t)l;
    __builtin_amdgcn_global_load_async_to_lds_b128(gp, lp, 0, 0);
}
__device__ __forceinline__ void wait_async0() {
#if __has_builtin(__builtin_amdgcn_s_wait_asynccnt)
    __builtin_amdgcn_s_wait_asynccnt(0);
#else
    asm volatile("s_wait_asynccnt 0" ::: "memory");
#endif
}
#endif

// Sizes
#define B_   8
#define CIN_ 512
#define HW_  3600
#define N_   28800          // B*HW
#define NT_  1800           // N_/16
#define NCLS_ 19
#define OUTPIX_ (8*19*480*480)   // 35020800
#define ATTEL_  (8*19*2304*36)   // 12607488

// ================= K0: pooling pyramid (p = 6,3,2,1) =================
__global__ __launch_bounds__(64) void k_pool(const float* __restrict__ F,
    float* __restrict__ P6, float* __restrict__ P3,
    float* __restrict__ P2, float* __restrict__ P1)
{
    __shared__ float s6[36];
    int bc = blockIdx.x;
    int t  = threadIdx.x;
    const float* f = F + (long long)bc * HW_;
    if (t < 36) {
        int by = t / 6, bx = t % 6;
        float s = 0.f;
        for (int y = 0; y < 10; ++y)
            for (int x = 0; x < 10; ++x)
                s += f[(by*10+y)*60 + bx*10+x];
        s *= 0.01f;
        P6[bc*36 + t] = s; s6[t] = s;
    }
    __syncthreads();
    if (t < 9) {
        int by = t/3, bx = t%3;
        float s = 0.25f*(s6[(2*by)*6+2*bx]   + s6[(2*by)*6+2*bx+1]
                       + s6[(2*by+1)*6+2*bx] + s6[(2*by+1)*6+2*bx+1]);
        P3[bc*9 + t] = s;
    }
    if (t < 4) {
        int by = t/2, bx = t%2;
        float s = 0.f;
        for (int y = 0; y < 3; ++y)
            for (int x = 0; x < 3; ++x)
                s += s6[(3*by+y)*6 + 3*bx+x];
        P2[bc*4 + t] = s * (1.0f/9.0f);
    }
    if (t == 0) {
        float s = 0.f;
        for (int i = 0; i < 36; ++i) s += s6[i];
        P1[bc] = s * (1.0f/36.0f);
    }
}

// ================= K1a: branch 1x1 convs + bias/BN + relu =================
__global__ __launch_bounds__(256) void k_branch_conv(
    const float* __restrict__ Wpsp, const float* __restrict__ b0,
    const float* __restrict__ gam, const float* __restrict__ bet,
    const float* __restrict__ mu,  const float* __restrict__ var,
    const float* __restrict__ P6, const float* __restrict__ P3,
    const float* __restrict__ P2, const float* __restrict__ P1,
    float* __restrict__ Z)
{
    int tid = blockIdx.x * 256 + threadIdx.x;
    if (tid >= 51200) return;
    int i, rem, psz, zb; const float* P;
    if (tid < 1024)      { i=0; rem=tid;        psz=1; P=P1; zb=0; }
    else if (tid < 5120) { i=1; rem=tid-1024;   psz=2; P=P2; zb=1024; }
    else if (tid < 14336){ i=2; rem=tid-5120;   psz=3; P=P3; zb=5120; }
    else                 { i=3; rem=tid-14336;  psz=6; P=P6; zb=14336; }
    int pp = psz * psz;
    int sp = rem % pp; int bj = rem / pp;
    int j = bj % 128;  int b  = bj / 128;
    const float* w  = Wpsp + (i*128 + j)*512;
    const float* pv = P + (long long)b*512*pp + sp;
    float acc = 0.f;
    for (int c = 0; c < 512; ++c) acc += w[c] * pv[c*pp];
    if (i == 0) acc += b0[j];
    else {
        int bi = (i-1)*128 + j;
        float sc = gam[bi] * rsqrtf(var[bi] + 1e-5f);
        acc = (acc - mu[bi]) * sc + bet[bi];
    }
    Z[zb + rem] = fmaxf(acc, 0.f);
}

// ====== K1p: convert+swizzle w_conv -> bf16 hi/lo in A-fragment order ======
// A block layout: block = (k>>5)*32 + (o>>4); within block (512 ushorts):
//   lane = (kl>>4)*16 + (o&15), elem = kl&15
__global__ __launch_bounds__(256) void k_prep_w(const float* __restrict__ W,
    unsigned short* __restrict__ Wfh, unsigned short* __restrict__ Wfl)
{
    int tid = blockIdx.x * 256 + threadIdx.x;     // 512*128
    if (tid >= 512*128) return;
    int o = tid >> 7;
    int k = (tid & 127) * 8;
    const f32x4* wp = (const f32x4*)(W + o*1024 + k);
    f32x4 w0 = wp[0], w1 = wp[1];
    float v[8] = {w0.x, w0.y, w0.z, w0.w, w1.x, w1.y, w1.z, w1.w};
    u32x4 qh, ql;
    split8(v, qh, ql);
    int kl    = k & 31;
    int block = (k >> 5) * 32 + (o >> 4);
    int lane  = (kl >> 4) * 16 + (o & 15);
    int j     = (kl >> 3) & 1;
    long long qi = (long long)block * 64 + lane * 2 + j;
    ((u32x4*)Wfh)[qi] = qh;
    ((u32x4*)Wfl)[qi] = ql;
}

// ====== K1b: branches upsample + concat feat -> bf16 hi/lo, B-frag order ====
// B block layout: block = (c>>4)*NT_ + (n>>4); within block (256 ushorts):
//   offset = (n&15)*16 + (c&15)
__global__ __launch_bounds__(256) void k_build_xcat(
    const float* __restrict__ F, const float* __restrict__ Z,
    unsigned short* __restrict__ Xfh, unsigned short* __restrict__ Xfl)
{
    long long tid = (long long)blockIdx.x * 256 + threadIdx.x;
    if (tid >= (long long)128 * N_) return;
    int n  = (int)(tid % N_);
    int cg = (int)(tid / N_);               // channel group of 8
    int b = n / HW_, hw = n % HW_;
    int h = hw / 60, w = hw % 60;
    float v[8];
    #pragma unroll
    for (int d = 0; d < 8; ++d) {
        int c = cg * 8 + d;
        float val;
        if (c >= 512) {
            val = F[((long long)(b*512 + (c-512)))*HW_ + hw];
        } else {
            int i = c >> 7, j = c & 127;
            const int psz_tab[4]  = {1, 2, 3, 6};
            const int zoff_tab[4] = {0, 1024, 5120, 14336};
            int p = psz_tab[i];
            const float* z = Z + zoff_tab[i] + (b*128 + j)*p*p;
            if (p == 1) val = z[0];
            else {
                float s = (float)(p-1) / 59.0f;
                float ys = h*s, xs = w*s;
                int y0 = (int)floorf(ys); int y1 = min(y0+1, p-1);
                int x0 = (int)floorf(xs); int x1 = min(x0+1, p-1);
                float wy = ys - (float)y0, wx = xs - (float)x0;
                float top = (1.f-wx)*z[y0*p+x0] + wx*z[y0*p+x1];
                float bot = (1.f-wx)*z[y1*p+x0] + wx*z[y1*p+x1];
                val = (1.f-wy)*top + wy*bot;
            }
        }
        v[d] = val;
    }
    u32x4 qh, ql;
    split8(v, qh, ql);
    int c0 = cg * 8;
    long long block = (long long)(c0 >> 4) * NT_ + (n >> 4);
    int j = (c0 >> 3) & 1;
    long long qi = block * 32 + (n & 15) * 2 + j;
    ((u32x4*)Xfh)[qi] = qh;
    ((u32x4*)Xfl)[qi] = ql;
}

// ================= K2: WMMA GEMM 512x28800x1024 + BN + ReLU ================
// C = W @ Xcat with bf16 hi/lo split: a*b ~= ah*bh + ah*bl + al*bh
__global__ __launch_bounds__(256) void k_gemm_bn_relu(
    const unsigned short* __restrict__ Wfh, const unsigned short* __restrict__ Wfl,
    const unsigned short* __restrict__ Xfh, const unsigned short* __restrict__ Xfl,
    const float* __restrict__ gamma, const float* __restrict__ beta,
    const float* __restrict__ mean,  const float* __restrict__ var,
    float* __restrict__ Y)            // [512][28800]
{
    __shared__ u32x4 sAh[256];        // [mi(4)][lane(32)][2]
    __shared__ u32x4 sAl[256];
    __shared__ u32x4 sBh[512];        // [t8(8)][lane(32)][2]
    __shared__ u32x4 sBl[512];

    const u32x4* WqH = (const u32x4*)Wfh;
    const u32x4* WqL = (const u32x4*)Wfl;
    const u32x4* XqH = (const u32x4*)Xfh;
    const u32x4* XqL = (const u32x4*)Xfl;

    const int tid  = threadIdx.x;
    const int lane = tid & 31;
    const int wid  = tid >> 5;
    const int mi   = wid & 3;       // M sub-tile (16 rows)
    const int nj   = wid >> 2;      // N half (64 cols)
    const int n0   = blockIdx.x * 128;
    const int o0   = blockIdx.y * 64;
    const int nt0  = n0 >> 4;

    v8f acc0 = {0,0,0,0,0,0,0,0};
    v8f acc1 = acc0, acc2 = acc0, acc3 = acc0;

    // B-stage thread mapping
    const int skh = tid >> 7;        // k half (16 k's)
    const int sr  = tid & 127;
    const int st8 = sr >> 4;         // n sub-tile
    const int snl = sr & 15;
    const int sl  = (st8*32 + skh*16 + snl)*2;

    for (int c0 = 0; c0 < 1024; c0 += 32) {
        long long abase = ((long long)(c0 >> 5) * 32 + (o0 >> 4)) * 64;
        long long bblk  = ((long long)((c0 >> 4) + skh) * NT_ + nt0 + st8) * 32 + snl*2;
#if USE_ASYNC_LDS
        // async global -> LDS (ASYNCcnt), 16B per lane per op
        gl2lds16(WqH + abase + tid, &sAh[tid]);
        gl2lds16(WqL + abase + tid, &sAl[tid]);
        gl2lds16(XqH + bblk,     &sBh[sl]);
        gl2lds16(XqH + bblk + 1, &sBh[sl + 1]);
        gl2lds16(XqL + bblk,     &sBl[sl]);
        gl2lds16(XqL + bblk + 1, &sBl[sl + 1]);
#else
        sAh[tid] = WqH[abase + tid];
        sAl[tid] = WqL[abase + tid];
        sBh[sl]     = XqH[bblk];
        sBh[sl + 1] = XqH[bblk + 1];
        sBl[sl]     = XqL[bblk];
        sBl[sl + 1] = XqL[bblk + 1];
#endif
        // prefetch next chunk's cachelines while this chunk computes
        if (c0 + 32 < 1024) {
            __builtin_prefetch((const void*)(WqH + abase + 2048 + tid), 0, 3);
            __builtin_prefetch((const void*)(WqL + abase + 2048 + tid), 0, 3);
            __builtin_prefetch((const void*)(XqH + bblk + 2*(long long)NT_*32), 0, 3);
            __builtin_prefetch((const void*)(XqL + bblk + 2*(long long)NT_*32), 0, 3);
        }
#if USE_ASYNC_LDS
        wait_async0();
#endif
        __syncthreads();

        BF16x16 fah, fal;
        {
            int ai = (mi*32 + lane)*2;
            fah.q[0] = sAh[ai]; fah.q[1] = sAh[ai+1];
            fal.q[0] = sAl[ai]; fal.q[1] = sAl[ai+1];
        }
        v8f* accp[4] = { &acc0, &acc1, &acc2, &acc3 };
        #pragma unroll
        for (int t = 0; t < 4; ++t) {
            BF16x16 fbh, fbl;
            int bi = ((nj*4 + t)*32 + lane)*2;
            fbh.q[0] = sBh[bi]; fbh.q[1] = sBh[bi+1];
            fbl.q[0] = sBl[bi]; fbl.q[1] = sBl[bi+1];
            v8f a = *accp[t];
            a = __builtin_amdgcn_wmma_f32_16x16x32_bf16(false, fah.v, false, fbh.v, (short)0, a, false, false);
            a = __builtin_amdgcn_wmma_f32_16x16x32_bf16(false, fah.v, false, fbl.v, (short)0, a, false, false);
            a = __builtin_amdgcn_wmma_f32_16x16x32_bf16(false, fal.v, false, fbh.v, (short)0, a, false, false);
            *accp[t] = a;
        }
        __syncthreads();
    }

    // epilogue: BN (eval) + ReLU, scatter per C/D layout
    F32x8 res[4]; res[0].v = acc0; res[1].v = acc1; res[2].v = acc2; res[3].v = acc3;
    #pragma unroll
    for (int t = 0; t < 4; ++t) {
        int nn = n0 + nj*64 + t*16 + (lane & 15);
        #pragma unroll
        for (int vv = 0; vv < 8; ++vv) {
            int o = o0 + mi*16 + vv + ((lane >> 4) ? 8 : 0);
            float sc = gamma[o] * rsqrtf(var[o] + 1e-5f);
            float sh = beta[o] - mean[o] * sc;
            float val = res[t].f[vv] * sc + sh;
            Y[(long long)o * N_ + nn] = fmaxf(val, 0.f);
        }
    }
}

// ================= K3: 3x3 conv 512->19 + bias =================
__global__ __launch_bounds__(256) void k_conv3x3(
    const float* __restrict__ X2,   // [512][28800]
    const float* __restrict__ Wseg, // [19][512][3][3]
    const float* __restrict__ bseg,
    float* __restrict__ X3)         // [8][19][3600]
{
    __shared__ float wsm[512*9];
    int cls = blockIdx.y, b = blockIdx.z;
    for (int i = threadIdx.x; i < 4608; i += 256)
        wsm[i] = Wseg[cls*4608 + i];
    __syncthreads();
    int pix = blockIdx.x * 256 + threadIdx.x;
    if (pix >= HW_) return;
    int h = pix / 60, w = pix % 60;
    float acc = bseg[cls];
    for (int c = 0; c < 512; ++c) {
        const float* xp = X2 + (long long)c * N_ + b * HW_;
        const float* wp = wsm + c * 9;
        #pragma unroll
        for (int dy = -1; dy <= 1; ++dy) {
            int y = h + dy;
            if ((unsigned)y >= 60u) continue;
            #pragma unroll
            for (int dx = -1; dx <= 1; ++dx) {
                int x = w + dx;
                if ((unsigned)x >= 60u) continue;
                acc += xp[y*60 + x] * wp[(dy+1)*3 + (dx+1)];
            }
        }
    }
    X3[((long long)b*NCLS_ + cls)*HW_ + pix] = acc;
}

// ================= K4: bilinear align-corners 60 -> 480 =================
__global__ __launch_bounds__(256) void k_upsample480(
    const float* __restrict__ X3, float* __restrict__ Out)
{
    long long idx = (long long)blockIdx.x * 256 + threadIdx.x;
    if (idx >= (long long)OUTPIX_) return;
    int w  = (int)(idx % 480);
    int h  = (int)((idx / 480) % 480);
    int bc = (int)(idx / (480*480));
    const float s = 59.0f / 479.0f;
    float ys = h * s, xs = w * s;
    int y0 = (int)floorf(ys); int y1 = min(y0+1, 59);
    int x0 = (int)floorf(xs); int x1 = min(x0+1, 59);
    float wy = ys - (float)y0, wx = xs - (float)x0;
    const float* src = X3 + (long long)bc * HW_;
    float top = (1.f-wx)*src[y0*60+x0] + wx*src[y0*60+x1];
    float bot = (1.f-wx)*src[y1*60+x0] + wx*src[y1*60+x1];
    Out[idx] = (1.f-wy)*top + wy*bot;
}

// ====== K5: per-pixel argmax -> one-hot 8x8 pooled -> unfold[b][c][36][100] ==
__global__ __launch_bounds__(64) void k_argmax_pool_unfold(
    const float* __restrict__ Xup, float* __restrict__ Unf)
{
    __shared__ int cnt[NCLS_];
    int bid = blockIdx.x;
    int b = bid / HW_, cell = bid % HW_;
    int cy = cell / 60, cx = cell % 60;
    int t = threadIdx.x;
    if (t < NCLS_) cnt[t] = 0;
    __syncthreads();
    int py = cy*8 + (t >> 3), px = cx*8 + (t & 7);
    const float* p = Xup + (((long long)b*NCLS_)*480 + py)*480 + px;
    float best = p[0]; int am = 0;
    for (int c = 1; c < NCLS_; ++c) {
        float v = p[(long long)c * 230400];
        if (v > best) { best = v; am = c; }
    }
    atomicAdd(&cnt[am], 1);
    __syncthreads();
    if (t < NCLS_) {
        int blk = (cy/10)*6 + (cx/10);
        int q   = (cy%10)*10 + (cx%10);
        Unf[(((long long)b*NCLS_ + t)*36 + blk)*100 + q] = (float)cnt[t] * (1.0f/64.0f);
    }
}

// ================= K7: 1/(count_nonzero + 1e-5) over k axis ==============
__global__ __launch_bounds__(256) void k_count_inv(
    const float* __restrict__ Attn, float* __restrict__ Inv)
{
    int idx = blockIdx.x * 256 + threadIdx.x;
    if (idx >= 8*19*2304) return;
    const float* a = Attn + (long long)idx * 36;
    int c = 0;
    for (int k = 0; k < 36; ++k) c += (a[k] != 0.0f);
    Inv[idx] = 1.0f / ((float)c + 1e-5f);
}

// ===== K8: att = (attn/cnt)@unf, fold, out = x*(1+att) in-place on d_out ====
__global__ __launch_bounds__(128) void k_fuse_att(
    const float* __restrict__ Attn, const float* __restrict__ Inv,
    const float* __restrict__ Unf,  float* __restrict__ Out)
{
    __shared__ float unfs[3600];
    __shared__ float ats[32*36];
    __shared__ float invs[32];
    int bc = blockIdx.x;
    int pg = blockIdx.y;
    int t  = threadIdx.x;
    for (int i = t; i < 3600; i += 128) unfs[i] = Unf[(long long)bc*3600 + i];
    int p0 = pg * 32;
    for (int i = t; i < 1152; i += 128) {
        int pp = i / 36, k = i % 36;
        ats[i] = Attn[((long long)bc*2304 + p0 + pp)*36 + k];
    }
    if (t < 32) invs[t] = Inv[(long long)bc*2304 + p0 + t];
    __syncthreads();
    for (int i = t; i < 3200; i += 128) {
        int pp = i / 100, q = i % 100;
        const float* ar = ats + pp*36;
        float s = 0.f;
        #pragma unroll
        for (int k = 0; k < 36; ++k) s += ar[k] * unfs[k*100 + q];
        s *= invs[pp];
        int p = p0 + pp;
        int h = (p/48)*10 + q/10;
        int w = (p%48)*10 + q%10;
        long long idx = (((long long)bc)*480 + h)*480 + w;
        float x = Out[idx];
        Out[idx] = x + s * x;
    }
}

// =========================== launch ===========================
extern "C" void kernel_launch(void* const* d_in, const int* in_sizes, int n_in,
                              void* d_out, int out_size, void* d_ws, size_t ws_size,
                              hipStream_t stream)
{
    const float* feat   = (const float*)d_in[0];
    const float* attn   = (const float*)d_in[1];
    const float* w_psp  = (const float*)d_in[2];
    const float* b_psp0 = (const float*)d_in[3];
    const float* pg     = (const float*)d_in[4];
    const float* pb     = (const float*)d_in[5];
    const float* pm     = (const float*)d_in[6];
    const float* pv     = (const float*)d_in[7];
    const float* w_conv = (const float*)d_in[8];
    const float* cg     = (const float*)d_in[9];
    const float* cb     = (const float*)d_in[10];
    const float* cm     = (const float*)d_in[11];
    const float* cv     = (const float*)d_in[12];
    const float* w_seg  = (const float*)d_in[13];
    const float* b_seg  = (const float*)d_in[14];
    float* out = (float*)d_out;

    float* ws = (float*)d_ws;
    float* P6   = ws;                     // 147456
    float* P3   = P6  + 147456;           // 36864
    float* P2   = P3  + 36864;            // 16384
    float* P1   = P2  + 16384;            // 4096
    float* Z    = P1  + 4096;             // 51200  (ends at 256000 floats)
    unsigned short* Wfh = (unsigned short*)(ws + 256000);   // 524288 ushorts
    unsigned short* Wfl = Wfh + 524288;
    unsigned short* Xfh = Wfl + 524288;                     // 29491200 ushorts
    unsigned short* Xfl = Xfh + (long long)29491200;
    float* X2   = (float*)(Xfl + (long long)29491200);      // 14745600 floats
    float* X3   = X2 + (long long)512*N_;                   // 547200
    float* Unf  = X3 + 547200;                              // 547200
    float* Inv  = Unf + 547200;                             // 350208

    k_pool<<<dim3(B_*CIN_), dim3(64), 0, stream>>>(feat, P6, P3, P2, P1);
    k_branch_conv<<<dim3((51200+255)/256), dim3(256), 0, stream>>>(
        w_psp, b_psp0, pg, pb, pm, pv, P6, P3, P2, P1, Z);
    k_prep_w<<<dim3(512*128/256), dim3(256), 0, stream>>>(w_conv, Wfh, Wfl);
    k_build_xcat<<<dim3((unsigned)(((long long)128*N_ + 255)/256)), dim3(256), 0, stream>>>(
        feat, Z, Xfh, Xfl);
    k_gemm_bn_relu<<<dim3(N_/128, 512/64), dim3(256), 0, stream>>>(
        Wfh, Wfl, Xfh, Xfl, cg, cb, cm, cv, X2);
    k_conv3x3<<<dim3((HW_+255)/256, NCLS_, B_), dim3(256), 0, stream>>>(
        X2, w_seg, b_seg, X3);
    k_upsample480<<<dim3((unsigned)(((long long)OUTPIX_ + 255)/256)), dim3(256), 0, stream>>>(
        X3, out);
    k_argmax_pool_unfold<<<dim3(B_*HW_), dim3(64), 0, stream>>>(out, Unf);
    k_count_inv<<<dim3((8*19*2304 + 255)/256), dim3(256), 0, stream>>>(attn, Inv);
    k_fuse_att<<<dim3(B_*NCLS_, 2304/32), dim3(128), 0, stream>>>(attn, Inv, Unf, out);
    (void)hipMemcpyAsync(out + (long long)OUTPIX_, attn, (size_t)ATTEL_ * sizeof(float),
                         hipMemcpyDeviceToDevice, stream);
}
